// MultiHeadCrossAttention_21114059227303
// MI455X (gfx1250) — matrix operand order repaired and checked by reading, not compiled, not run on hardware
//
#include <hip/hip_runtime.h>

typedef unsigned short u16;
typedef __attribute__((ext_vector_type(16))) __bf16 v16bf;
typedef __attribute__((ext_vector_type(8)))  float  v8f;
typedef int i32x4 __attribute__((__vector_size__(16)));

#define IN_DIM 256
#define NHEADS 8
#define HDIM   32
#define BATCH  2
#define NTOK   4096          // 64*64 tokens
#define HN     (NTOK * HDIM) // per-(b,h) q/k/v plane: 131072 elems

union Frag {
  v16bf v;
  uint4 q[2];
  u16   u[16];
};

__device__ __forceinline__ u16 f2bf(float f) {
  union { __bf16 b; u16 u; } cv;
  cv.b = (__bf16)f;
  return cv.u;
}

// A fragment (16x32 bf16, row-major source, K contiguous).
// row = r0 + (lane&15); lanes 0-15: K {0-7,16-23}; lanes 16-31: K {8-15,24-31}
__device__ __forceinline__ v16bf load_a(const u16* base, int rowStride, int r0,
                                        int k0, int lane) {
  const int hi = lane >> 4;
  const u16* p = base + (size_t)(r0 + (lane & 15)) * rowStride + k0;
  Frag f;
  f.q[0] = *(const uint4*)(p + hi * 8);
  f.q[1] = *(const uint4*)(p + 16 + hi * 8);
  return f.v;
}

// B fragment (32x16 bf16) loaded from B^T stored row-major [N][K].
// column n = n0 + (lane&15); lanes 0-15: K {0-7,16-23}; lanes 16-31: K {8-15,24-31}
__device__ __forceinline__ v16bf load_bt(const u16* base, int rowStride, int n0,
                                         int k0, int lane) {
  const int hi = lane >> 4;
  const u16* p = base + (size_t)(n0 + (lane & 15)) * rowStride + k0;
  Frag f;
  f.q[0] = *(const uint4*)(p + hi * 8);
  f.q[1] = *(const uint4*)(p + 16 + hi * 8);
  return f.v;
}

__device__ __forceinline__ v8f wmma_bf16(v16bf a, v16bf b, v8f c) {
  // (neg_a, A, neg_b, B, c_mod, C, reuse_a, reuse_b)
  return __builtin_amdgcn_wmma_f32_16x16x32_bf16(false, a, false, b, (short)0, c,
                                                 false, false);
}

// --- async global->LDS copy (gfx1250 path), 16 bytes per lane ---------------
__device__ __forceinline__ void async_cp16(const u16* g, u16* l) {
#if __has_builtin(__builtin_amdgcn_global_load_async_to_lds_b128)
  __builtin_amdgcn_global_load_async_to_lds_b128(
      (__attribute__((address_space(1))) i32x4*)(u16*)g,
      (__attribute__((address_space(3))) i32x4*)l, 0, 0);
#else
  *(uint4*)l = *(const uint4*)g;  // synchronous fallback
#endif
}

__device__ __forceinline__ void wait_async0() {
#if __has_builtin(__builtin_amdgcn_s_wait_asynccnt)
  __builtin_amdgcn_s_wait_asynccnt(0);
#elif __has_builtin(__builtin_amdgcn_global_load_async_to_lds_b128)
  asm volatile("s_wait_asynccnt 0" ::: "memory");
#endif
}

// ---------------------------------------------------------------------------
// Kernel 0: stage fp32 weights as bf16 in workspace (row-major [d][c] kept).
// ---------------------------------------------------------------------------
__global__ void __launch_bounds__(256) cvt_w_kernel(
    const float* __restrict__ Wq, const float* __restrict__ Wk,
    const float* __restrict__ Wv, const float* __restrict__ Wo,
    u16* __restrict__ wsW) {
  int i = blockIdx.x * 256 + threadIdx.x;       // 0 .. 4*65536-1
  int sel = i >> 16, j = i & 65535;
  const float* s = (sel == 0) ? Wq : (sel == 1) ? Wk : (sel == 2) ? Wv : Wo;
  wsW[i] = f2bf(s[j]);
}

// ---------------------------------------------------------------------------
// Kernel 1: QKV projections. One WG per (b, 64-token tile).
// Stages channel-transposed x tile in LDS (bf16), 8 waves do 16x16x32 WMMAs.
// Q scaled by 1/sqrt(32), stored [b][h][n][dh]; K same; V stored transposed
// [b][h][dh][n] so the attention kernel's B-fragments are contiguous in n.
// ---------------------------------------------------------------------------
#define LSTR 264  // padded row stride (bf16 elems); 528B = 33*16 keeps loads aligned

__global__ void __launch_bounds__(256) qkv_proj_kernel(
    const float* __restrict__ tgt, const float* __restrict__ srcx,
    const u16* __restrict__ wsW, u16* __restrict__ wsQ, u16* __restrict__ wsK,
    u16* __restrict__ wsVT) {
  __shared__ __align__(16) u16 ldsX[64 * LSTR];
  const int tid = threadIdx.x;
  const int b = blockIdx.x >> 6;
  const int n0 = (blockIdx.x & 63) * 64;
  const int w = tid >> 5, lane = tid & 31, hi = lane >> 4;
  const int r0 = (w & 3) * 16;       // row sub-tile within the 64 rows
  const int dbase = (w >> 2) * 128;  // column half

  const u16* Wq = wsW;
  const u16* Wk = wsW + 65536;
  const u16* Wv = wsW + 2 * 65536;

  // ---- stage tgt tile: lds[n_local][c] = bf16(tgt[b][c][n0+n_local]) ----
  {
    const float* xp = tgt + (size_t)b * IN_DIM * NTOK;
    for (int i = tid; i < IN_DIM * 16; i += 256) {
      int c = i >> 4, j4 = (i & 15) * 4;
      float4 f = *(const float4*)(xp + (size_t)c * NTOK + n0 + j4);
      ldsX[(j4 + 0) * LSTR + c] = f2bf(f.x);
      ldsX[(j4 + 1) * LSTR + c] = f2bf(f.y);
      ldsX[(j4 + 2) * LSTR + c] = f2bf(f.z);
      ldsX[(j4 + 3) * LSTR + c] = f2bf(f.w);
    }
  }
  __syncthreads();

  // ---- Q = tgt_t @ Wq^T, scaled by 1/sqrt(HDIM) ----
  const float qscale = 0.17677669529663687f;
  for (int dt = 0; dt < 8; ++dt) {
    int d0 = dbase + dt * 16;
    v8f acc = {0.f, 0.f, 0.f, 0.f, 0.f, 0.f, 0.f, 0.f};
#pragma unroll
    for (int ks = 0; ks < 8; ++ks) {
      v16bf a  = load_a(ldsX, LSTR, r0, ks * 32, lane);
      v16bf bm = load_bt(Wq, IN_DIM, d0, ks * 32, lane);
      acc = wmma_bf16(a, bm, acc);
    }
    int d = d0 + (lane & 15);
    int h = d >> 5, dh = d & 31;
    u16* qp = wsQ + (size_t)(b * NHEADS + h) * HN;
#pragma unroll
    for (int v = 0; v < 8; ++v) {
      int n = n0 + r0 + v + 8 * hi;
      qp[n * HDIM + dh] = f2bf(acc[v] * qscale);
    }
  }
  __syncthreads();

  // ---- restage with src tile ----
  {
    const float* xp = srcx + (size_t)b * IN_DIM * NTOK;
    for (int i = tid; i < IN_DIM * 16; i += 256) {
      int c = i >> 4, j4 = (i & 15) * 4;
      float4 f = *(const float4*)(xp + (size_t)c * NTOK + n0 + j4);
      ldsX[(j4 + 0) * LSTR + c] = f2bf(f.x);
      ldsX[(j4 + 1) * LSTR + c] = f2bf(f.y);
      ldsX[(j4 + 2) * LSTR + c] = f2bf(f.z);
      ldsX[(j4 + 3) * LSTR + c] = f2bf(f.w);
    }
  }
  __syncthreads();

  // ---- K = src_t @ Wk^T ----
  for (int dt = 0; dt < 8; ++dt) {
    int d0 = dbase + dt * 16;
    v8f acc = {0.f, 0.f, 0.f, 0.f, 0.f, 0.f, 0.f, 0.f};
#pragma unroll
    for (int ks = 0; ks < 8; ++ks) {
      v16bf a  = load_a(ldsX, LSTR, r0, ks * 32, lane);
      v16bf bm = load_bt(Wk, IN_DIM, d0, ks * 32, lane);
      acc = wmma_bf16(a, bm, acc);
    }
    int d = d0 + (lane & 15);
    int h = d >> 5, dh = d & 31;
    u16* kp = wsK + (size_t)(b * NHEADS + h) * HN;
#pragma unroll
    for (int v = 0; v < 8; ++v) {
      int n = n0 + r0 + v + 8 * hi;
      kp[n * HDIM + dh] = f2bf(acc[v]);
    }
  }

  // ---- V = src_t @ Wv^T, stored transposed [dh][n] ----
  for (int dt = 0; dt < 8; ++dt) {
    int d0 = dbase + dt * 16;
    v8f acc = {0.f, 0.f, 0.f, 0.f, 0.f, 0.f, 0.f, 0.f};
#pragma unroll
    for (int ks = 0; ks < 8; ++ks) {
      v16bf a  = load_a(ldsX, LSTR, r0, ks * 32, lane);
      v16bf bm = load_bt(Wv, IN_DIM, d0, ks * 32, lane);
      acc = wmma_bf16(a, bm, acc);
    }
    int d = d0 + (lane & 15);
    int h = d >> 5, dh = d & 31;
    u16* vp = wsVT + (size_t)(b * NHEADS + h) * HN;
#pragma unroll
    for (int v = 0; v < 8; ++v) {
      int n = n0 + r0 + v + 8 * hi;
      vp[dh * NTOK + n] = f2bf(acc[v]);
    }
  }
}

// ---------------------------------------------------------------------------
// Kernel 2: flash attention with async double-buffered K/V staging in LDS.
// WG = 8 waves sharing one (b,h); each wave owns 16 query rows; Q fragment
// stays in registers across the whole 4096-key loop. K (64x32) and V^T
// (32x64) tiles are staged once per WG via global_load_async_to_lds_b128
// (ASYNCcnt) and double-buffered so the copy of block i+1 overlaps block i's
// WMMAs + online softmax. Residual (+tgt_t) fused into the epilogue.
// ---------------------------------------------------------------------------
__global__ void __launch_bounds__(256) attn_kernel(
    const u16* __restrict__ wsQ, const u16* __restrict__ wsK,
    const u16* __restrict__ wsVT, const float* __restrict__ tgt,
    u16* __restrict__ wsA) {
  __shared__ __align__(16) u16 ldsK[2][64 * 32];   // 2 x 4KB
  __shared__ __align__(16) u16 ldsV[2][32 * 64];   // 2 x 4KB
  __shared__ __align__(16) u16 ldsP[8 * 16 * 72];  // wave-private 16x64 P tiles

  const int tid = threadIdx.x;
  const int bh = blockIdx.x >> 5;  // 0..15 = b*8+h
  const int qt = blockIdx.x & 31;  // 128-row query tile
  const int w = tid >> 5, lane = tid & 31, hi = lane >> 4;
  const int qr0 = qt * 128 + w * 16;

  const u16* qb = wsQ + (size_t)bh * HN;
  const u16* kbg = wsK + (size_t)bh * HN;
  const u16* vbg = wsVT + (size_t)bh * HN;
  u16* myP = ldsP + w * (16 * 72);

  // staging assignment: one 16B chunk per thread per tile
  const int krow = tid >> 2, kch = (tid & 3) * 8;  // K: 64 rows x 4 chunks
  const int vrow = tid >> 3, vch = (tid & 7) * 8;  // V: 32 rows x 8 chunks

  v16bf qa = load_a(qb, HDIM, qr0, 0, lane);  // 16 q-rows x full head dim
  v8f o0 = {0.f, 0.f, 0.f, 0.f, 0.f, 0.f, 0.f, 0.f};
  v8f o1 = {0.f, 0.f, 0.f, 0.f, 0.f, 0.f, 0.f, 0.f};
  float mrun[8], lrun[8];
#pragma unroll
  for (int v = 0; v < 8; ++v) { mrun[v] = -3.0e38f; lrun[v] = 0.f; }

  // prologue stage of block 0
  {
    const int kn0 = 0;
    async_cp16(kbg + (size_t)(kn0 + krow) * HDIM + kch, &ldsK[0][krow * 32 + kch]);
    async_cp16(vbg + (size_t)vrow * NTOK + kn0 + vch, &ldsV[0][vrow * 64 + vch]);
  }

  for (int blk = 0; blk < NTOK / 64; ++blk) {
    wait_async0();      // this wave's pending copies (block `blk`) have landed
    __syncthreads();    // ...and every other wave's too

    if (blk + 1 < NTOK / 64) {
      const int kn0 = (blk + 1) * 64;
      const int nb = (blk + 1) & 1;
      async_cp16(kbg + (size_t)(kn0 + krow) * HDIM + kch,
                 &ldsK[nb][krow * 32 + kch]);
      async_cp16(vbg + (size_t)vrow * NTOK + kn0 + vch,
                 &ldsV[nb][vrow * 64 + vch]);
    }

    const u16* kbuf = ldsK[blk & 1];
    const u16* vbuf = ldsV[blk & 1];

    // scores for 64 keys: 4 WMMA tiles
    v16bf kf0 = load_bt(kbuf, HDIM, 0, 0, lane);
    v16bf kf1 = load_bt(kbuf, HDIM, 16, 0, lane);
    v16bf kf2 = load_bt(kbuf, HDIM, 32, 0, lane);
    v16bf kf3 = load_bt(kbuf, HDIM, 48, 0, lane);
    v8f z = {0.f, 0.f, 0.f, 0.f, 0.f, 0.f, 0.f, 0.f};
    v8f s0 = wmma_bf16(qa, kf0, z);
    v8f s1 = wmma_bf16(qa, kf1, z);
    v8f s2 = wmma_bf16(qa, kf2, z);
    v8f s3 = wmma_bf16(qa, kf3, z);

    // online softmax over the 64 new columns (one reduction per 64 keys)
#pragma unroll
    for (int v = 0; v < 8; ++v) {
      float t = fmaxf(fmaxf(s0[v], s1[v]), fmaxf(s2[v], s3[v]));
      t = fmaxf(t, __shfl_xor(t, 1, 16));
      t = fmaxf(t, __shfl_xor(t, 2, 16));
      t = fmaxf(t, __shfl_xor(t, 4, 16));
      t = fmaxf(t, __shfl_xor(t, 8, 16));
      float nm = fmaxf(mrun[v], t);
      float sc = __expf(mrun[v] - nm);
      float e0 = __expf(s0[v] - nm);
      float e1 = __expf(s1[v] - nm);
      float e2 = __expf(s2[v] - nm);
      float e3 = __expf(s3[v] - nm);
      float rs = (e0 + e1) + (e2 + e3);
      rs += __shfl_xor(rs, 1, 16);
      rs += __shfl_xor(rs, 2, 16);
      rs += __shfl_xor(rs, 4, 16);
      rs += __shfl_xor(rs, 8, 16);
      lrun[v] = lrun[v] * sc + rs;
      mrun[v] = nm;
      o0[v] *= sc;
      o1[v] *= sc;
      int m = v + 8 * hi;
      myP[m * 72 + (lane & 15)] = f2bf(e0);  // C-frag layout -> row-major P
      myP[m * 72 + 16 + (lane & 15)] = f2bf(e1);
      myP[m * 72 + 32 + (lane & 15)] = f2bf(e2);
      myP[m * 72 + 48 + (lane & 15)] = f2bf(e3);
    }
    // DS ops from one wave are in-order; P tile is wave-private -> no barrier.
    v16bf pa0 = load_a(myP, 72, 0, 0, lane);
    v16bf pa1 = load_a(myP, 72, 0, 32, lane);
    v16bf vf00 = load_bt(vbuf, 64, 0, 0, lane);   // dh 0-15,  k 0-31
    v16bf vf01 = load_bt(vbuf, 64, 0, 32, lane);  // dh 0-15,  k 32-63
    v16bf vf10 = load_bt(vbuf, 64, 16, 0, lane);  // dh 16-31, k 0-31
    v16bf vf11 = load_bt(vbuf, 64, 16, 32, lane); // dh 16-31, k 32-63
    o0 = wmma_bf16(pa0, vf00, o0);
    o0 = wmma_bf16(pa1, vf01, o0);
    o1 = wmma_bf16(pa0, vf10, o1);
    o1 = wmma_bf16(pa1, vf11, o1);
    // buffer (blk&1) is not overwritten until iteration blk+1's stage of
    // blk+2, which happens after the next top-of-loop barrier.
  }

  // epilogue: O/l + residual tgt_t, bf16 store [b][n][c]
  const int b = bh >> 3, h = bh & 7;
  const float* tp = tgt + (size_t)b * IN_DIM * NTOK;
#pragma unroll
  for (int v = 0; v < 8; ++v) {
    int n = qr0 + v + 8 * hi;
    float inv = 1.0f / lrun[v];
    int c0 = h * HDIM + (lane & 15);
    float r0v = o0[v] * inv + tp[(size_t)c0 * NTOK + n];
    float r1v = o1[v] * inv + tp[(size_t)(c0 + 16) * NTOK + n];
    u16* ap = wsA + (size_t)(b * NTOK + n) * IN_DIM;
    ap[c0] = f2bf(r0v);
    ap[c0 + 16] = f2bf(r1v);
  }
}

// ---------------------------------------------------------------------------
// Kernel 3: output projection: out = (weighted + tgt_t) @ Wo^T, fp32 output.
// ---------------------------------------------------------------------------
__global__ void __launch_bounds__(256) out_proj_kernel(
    const u16* __restrict__ wsA, const u16* __restrict__ wsW,
    float* __restrict__ out) {
  const int tid = threadIdx.x;
  const int bn0 = blockIdx.x * 64;  // global row over B*N
  const int w = tid >> 5, lane = tid & 31, hi = lane >> 4;
  const int r0 = bn0 + (w & 3) * 16;
  const int dbase = (w >> 2) * 128;
  const u16* Wo = wsW + 3 * 65536;

  for (int dt = 0; dt < 8; ++dt) {
    int d0 = dbase + dt * 16;
    v8f acc = {0.f, 0.f, 0.f, 0.f, 0.f, 0.f, 0.f, 0.f};
#pragma unroll
    for (int ks = 0; ks < 8; ++ks) {
      v16bf a  = load_a(wsA, IN_DIM, r0, ks * 32, lane);
      v16bf bm = load_bt(Wo, IN_DIM, d0, ks * 32, lane);
      acc = wmma_bf16(a, bm, acc);
    }
    int d = d0 + (lane & 15);
#pragma unroll
    for (int v = 0; v < 8; ++v) {
      out[(size_t)(r0 + v + 8 * hi) * IN_DIM + d] = acc[v];
    }
  }
}

// ---------------------------------------------------------------------------
extern "C" void kernel_launch(void* const* d_in, const int* in_sizes, int n_in,
                              void* d_out, int out_size, void* d_ws,
                              size_t ws_size, hipStream_t stream) {
  (void)in_sizes; (void)n_in; (void)out_size; (void)ws_size;
  const float* tgt = (const float*)d_in[0];
  const float* src = (const float*)d_in[1];
  const float* Wq = (const float*)d_in[2];
  const float* Wk = (const float*)d_in[3];
  const float* Wv = (const float*)d_in[4];
  const float* Wo = (const float*)d_in[5];

  u16* wsW  = (u16*)d_ws;                 // 4 * 65536 bf16 weights
  u16* wsQ  = wsW + 4 * 65536;            // [B][H][N][32] bf16
  u16* wsK  = wsQ + BATCH * NHEADS * HN;
  u16* wsVT = wsK + BATCH * NHEADS * HN;  // [B][H][32][N] bf16
  u16* wsA  = wsVT + BATCH * NHEADS * HN; // [B][N][C] bf16 (attn + residual)
  float* out = (float*)d_out;

  hipLaunchKernelGGL(cvt_w_kernel, dim3(1024), dim3(256), 0, stream,
                     Wq, Wk, Wv, Wo, wsW);
  hipLaunchKernelGGL(qkv_proj_kernel, dim3(BATCH * (NTOK / 64)), dim3(256), 0,
                     stream, tgt, src, wsW, wsQ, wsK, wsVT);
  hipLaunchKernelGGL(attn_kernel, dim3(BATCH * NHEADS * (NTOK / 128)), dim3(256),
                     0, stream, wsQ, wsK, wsVT, tgt, wsA);
  hipLaunchKernelGGL(out_proj_kernel, dim3(BATCH * NTOK / 64), dim3(256), 0,
                     stream, wsA, wsW, out);
}
